// Kernel_Conv_19885698580512
// MI455X (gfx1250) — matrix operand order repaired and checked by reading, not compiled, not run on hardware
//
#include <hip/hip_runtime.h>
#include <hip/hip_bf16.h>

typedef __bf16 bf16;
typedef __attribute__((ext_vector_type(8)))  __bf16 v8bf;
typedef __attribute__((ext_vector_type(16))) __bf16 v16bf;
typedef __attribute__((ext_vector_type(8)))  float  v8f;
typedef __attribute__((ext_vector_type(4)))  unsigned int u32x4;
typedef __attribute__((ext_vector_type(8)))  int i32x8;
typedef __attribute__((ext_vector_type(4)))  int i32x4;

// Problem constants (hardcoded from the reference)
#define BATCH 64
#define CIN   128
#define OC    128
#define HW    80
#define POOL  5      // k+2
#define KS    3
#define NPOS  (HW*HW)          // 6400
#define DYNW_ELEMS ((size_t)BATCH*9*OC*CIN)          // 9,437,184 bf16
#define DYNW_BYTES (DYNW_ELEMS*2)                    // 18,874,368 B

static __device__ __forceinline__ bf16 f2bf(float f) { return (bf16)f; }

// ---------------------------------------------------------------------------
// Kernel 1: adaptive avg pool 80x80 -> 5x5 (each cell = mean of 16x16 block).
// One wave (32 lanes) per output cell, 8 elements/lane, shfl reduction.
// ---------------------------------------------------------------------------
__global__ void pool_kernel(const float* __restrict__ x, float* __restrict__ avg) {
    int cc   = blockIdx.x * 8 + (threadIdx.x >> 5);   // cell id 0..204799
    int lane = threadIdx.x & 31;
    int b    = cc / (CIN * POOL * POOL);
    int rem  = cc % (CIN * POOL * POOL);
    int ch   = rem / (POOL * POOL);
    int cell = rem % (POOL * POOL);
    int oy = cell / POOL, ox = cell % POOL;

    const float* base = x + ((size_t)(b * CIN + ch)) * NPOS;
    float sum = 0.f;
#pragma unroll
    for (int i = 0; i < 8; ++i) {
        int e = lane + i * 32;                 // 0..255 within the 16x16 block
        int ry = oy * 16 + (e >> 4);
        int rx = ox * 16 + (e & 15);
        sum += base[ry * HW + rx];
    }
#pragma unroll
    for (int off = 16; off >= 1; off >>= 1)
        sum += __shfl_xor(sum, off, 32);
    if (lane == 0) avg[cc] = sum * (1.0f / 256.0f);
}

// ---------------------------------------------------------------------------
// Kernel 2: attention (3x3 VALID conv on 5x5 pooled maps + bias + sigmoid)
// fused with dyn_w = weight * att. One block per (b, o), thread = c.
// Output layout: dynW[b][p=kh*3+kw][o][c] as bf16 (contiguous GEMM A panels).
// ---------------------------------------------------------------------------
__global__ void att_kernel(const float* __restrict__ avg,
                           const float* __restrict__ weight,
                           const float* __restrict__ w_att,
                           const float* __restrict__ b_att,
                           bf16* __restrict__ dynW) {
    int b = blockIdx.x >> 7;
    int o = blockIdx.x & 127;
    int c = threadIdx.x;

    __shared__ float sAvg[POOL * POOL];
    if (threadIdx.x < POOL * POOL)
        sAvg[threadIdx.x] = avg[((size_t)(b * CIN + o)) * (POOL * POOL) + threadIdx.x];
    __syncthreads();

    int oc = o * CIN + c;                       // channel of grouped conv
    float wa[9];
#pragma unroll
    for (int t = 0; t < 9; ++t) wa[t] = w_att[(size_t)oc * 9 + t];
    float bias = b_att[oc];

#pragma unroll
    for (int kh = 0; kh < KS; ++kh) {
#pragma unroll
        for (int kw = 0; kw < KS; ++kw) {
            float s = bias;
#pragma unroll
            for (int r = 0; r < 3; ++r)
#pragma unroll
                for (int q = 0; q < 3; ++q)
                    s += wa[r * 3 + q] * sAvg[(kh + r) * POOL + (kw + q)];
            float att = 1.0f / (1.0f + __expf(-s));
            float dw  = weight[(size_t)oc * 9 + kh * 3 + kw] * att;
            int p = kh * 3 + kw;
            dynW[(((size_t)b * 9 + p) * OC + o) * CIN + c] = f2bf(dw);
        }
    }
}

// ---------------------------------------------------------------------------
// Kernel 3: per-sample implicit-GEMM conv via WMMA bf16, with the per-shift
// A panel (dynW[b][p], 128x128 bf16 = 32KB) staged into double-buffered LDS
// by the Tensor Data Mover (async, TENSORcnt), overlapped with compute.
// Block = (sample b, 8x8 spatial tile). 256 threads = 8 waves.
// M=128 (o, 16 rows/wave), N=64 (4 16-col tiles/wave), K = 9 shifts x 128 ch.
// ---------------------------------------------------------------------------
#define ROWP 136                                 // padded LDS row (bf16 elems)

// Issue a TDM 2-D tile load: dynW panel (128 rows x 128 bf16, row=256B) ->
// LDS with hardware row padding (64 DWORDs data + 4 DWORDs pad = 272B rows).
static __device__ __forceinline__ void tdm_load_panel(const bf16* gsrc,
                                                      unsigned lds_addr) {
    unsigned long long ga = (unsigned long long)(uintptr_t)gsrc;
    u32x4 g0;
    g0[0] = 1u;                                   // count=1, user descriptor
    g0[1] = lds_addr;                             // LDS byte address
    g0[2] = (unsigned)(ga & 0xFFFFFFFFull);       // global_addr[31:0]
    g0[3] = (unsigned)((ga >> 32) & 0x01FFFFFFull) | (2u << 30); // [56:32]|type=2

    i32x8 g1;
    g1[0] = (1 << 16)        // data_size = 1 -> 2 bytes
          | (1 << 20)        // pad_enable
          | (5 << 22)        // pad_interval code 5 = 64 DWORDs (256B row)
          | (3 << 25);       // pad_amount  code 3 = 4 DWORDs (16B pad)
    g1[1] = (int)(128u << 16);       // tensor_dim0[15:0]=128 (atomic addr 0)
    g1[2] = (int)(128u << 16);       // tensor_dim0 hi=0 | tensor_dim1 lo=128
    g1[3] = (int)(128u << 16);       // tensor_dim1 hi=0 | tile_dim0=128
    g1[4] = 128;                     // tile_dim1=128, tile_dim2=0
    g1[5] = 128;                     // tensor_dim0_stride = 128 elems
    g1[6] = (int)(16384u << 16);     // stride0 hi=0 | tensor_dim1_stride lo=16384
    g1[7] = 0;                       // tensor_dim1_stride hi

    i32x4 z4 = {0, 0, 0, 0};         // groups 2/3 unused (2-D tensor)
    i32x8 z8 = {0, 0, 0, 0, 0, 0, 0, 0};
    __builtin_amdgcn_tensor_load_to_lds(g0, g1, z4, z4, z8, 0);
}

__global__ void conv_kernel(const float* __restrict__ x,
                            const bf16* __restrict__ dynW,
                            float* __restrict__ out) {
    __shared__ bf16 sA[2][OC * ROWP];            // A panels: 2 x 34,816 B
    __shared__ bf16 sX[100 * ROWP];              // x patch: [pos(10x10)][c]

    int tid  = threadIdx.x;
    int wave = tid >> 5;
    int lane = tid & 31;
    int l15  = lane & 15;
    int lhalf = (lane < 16) ? 0 : 1;

    int b    = blockIdx.x / 100;
    int tile = blockIdx.x % 100;
    int y0   = (tile / 10) * 8;
    int x0   = (tile % 10) * 8;

    const bf16* gA = dynW + (size_t)b * 9 * (OC * CIN);

    // Prefetch A panel for shift p=0 via TDM (wave 0 only; DMA is per-wave).
    if (wave == 0)
        tdm_load_panel(gA, (unsigned)(uintptr_t)(void*)&sA[0][0]);

    // ---- stage x patch (128 ch x 10x10 halo), fp32 -> bf16, zero-padded ----
    const float* xb = x + (size_t)b * CIN * NPOS;
    for (int i = tid; i < CIN * 100; i += 256) {
        int c   = i / 100;
        int pos = i % 100;
        int gy  = y0 - 1 + pos / 10;
        int gx  = x0 - 1 + pos % 10;
        float v = (gy >= 0 && gy < HW && gx >= 0 && gx < HW)
                    ? xb[(size_t)c * NPOS + gy * HW + gx] : 0.0f;
        sX[pos * ROWP + c] = f2bf(v);
    }

    v8f acc[4];
#pragma unroll
    for (int i = 0; i < 4; ++i) acc[i] = (v8f)(0.0f);

    for (int p = 0; p < 9; ++p) {
        int kh = p / 3, kw = p % 3;
        int cur = p & 1;

        if (wave == 0)
            __builtin_amdgcn_s_wait_tensorcnt(0);   // panel p landed in LDS
        __syncthreads();   // panel p visible to all; all waves done with p-1

        // Prefetch panel p+1 into the other buffer while computing p.
        if (p < 8 && wave == 0)
            tdm_load_panel(gA + (size_t)(p + 1) * (OC * CIN),
                           (unsigned)(uintptr_t)(void*)&sA[cur ^ 1][0]);

        const bf16* As = &sA[cur][0];
#pragma unroll
        for (int ks = 0; ks < 4; ++ks) {
            int c0 = ks * 32;

            // A fragment: lane l -> row o = wave*16 + l%16,
            // K halves per ISA 16-bit A layout (lo: K 0..7 / 8..15, hi: +16)
            const bf16* ap = &As[(wave * 16 + l15) * ROWP + c0 + lhalf * 8];
            v8bf alo = *(const v8bf*)ap;
            v8bf ahi = *(const v8bf*)(ap + 16);
            v16bf afrag;
#pragma unroll
            for (int i = 0; i < 8; ++i) { afrag[i] = alo[i]; afrag[i + 8] = ahi[i]; }

#pragma unroll
            for (int nt = 0; nt < 4; ++nt) {
                int n  = nt * 16 + l15;
                int pos = (((n >> 3) + kh) * 10) + ((n & 7) + kw);
                const bf16* bp = &sX[pos * ROWP + c0 + lhalf * 8];
                v8bf blo = *(const v8bf*)bp;
                v8bf bhi = *(const v8bf*)(bp + 16);
                v16bf bfrag;
#pragma unroll
                for (int i = 0; i < 8; ++i) { bfrag[i] = blo[i]; bfrag[i + 8] = bhi[i]; }

                acc[nt] = __builtin_amdgcn_wmma_f32_16x16x32_bf16(
                    false, afrag, false, bfrag, (short)0, acc[nt], false, false);
            }
        }
    }

    // ---- writeback: C layout lane->N=l%16, VGPR r -> M = r + 8*(l>=16) ----
#pragma unroll
    for (int nt = 0; nt < 4; ++nt) {
        int n  = nt * 16 + l15;
        int gy = y0 + (n >> 3);
        int gx = x0 + (n & 7);
#pragma unroll
        for (int r = 0; r < 8; ++r) {
            int o = wave * 16 + r + lhalf * 8;
            out[(((size_t)b * OC + o) * HW + gy) * HW + gx] = acc[nt][r];
        }
    }
}

// ---------------------------------------------------------------------------
extern "C" void kernel_launch(void* const* d_in, const int* in_sizes, int n_in,
                              void* d_out, int out_size, void* d_ws, size_t ws_size,
                              hipStream_t stream) {
    const float* x      = (const float*)d_in[0];
    const float* weight = (const float*)d_in[1];
    const float* w_att  = (const float*)d_in[2];
    const float* b_att  = (const float*)d_in[3];
    float* out = (float*)d_out;

    bf16*  dynW = (bf16*)d_ws;
    float* avg  = (float*)((char*)d_ws + DYNW_BYTES);

    pool_kernel<<<BATCH * CIN * POOL * POOL / 8, 256, 0, stream>>>(x, avg);
    att_kernel <<<BATCH * OC, CIN, 0, stream>>>(avg, weight, w_att, b_att, dynW);
    conv_kernel<<<BATCH * 100, 256, 0, stream>>>(x, dynW, out);
}
